// SwitchLinear_52793738003186
// MI455X (gfx1250) — matrix operand order repaired and checked by reading, not compiled, not run on hardware
//
#include <hip/hip_runtime.h>
#include <hip/hip_bf16.h>
#include <stdint.h>

#define N_TOK  8192
#define D_IN   2048
#define D_OUTN 8192
#define NEXP   8

#define BM 128
#define BN 128
#define BK 32
#define KCHUNKS (D_IN / BK)
#define THREADS 256
#define PITCH 20    // u32 words per LDS row (16 data + pad, rows 16B-aligned, conflict-light)

typedef __attribute__((ext_vector_type(16))) __bf16 v16bf;
typedef __attribute__((ext_vector_type(2)))  __bf16 v2bf;
typedef __attribute__((ext_vector_type(8)))  float  v8f;

// ---------------- prep kernels (expert sort) ----------------
// ws layout (ints): [0..7] counts, [8..15] offsets, [16..23] cursors, [32..32+N) token list
__global__ void sl_zero(int* wsi) {
    if (threadIdx.x < 24) wsi[threadIdx.x] = 0;
}
__global__ void sl_hist(const int* __restrict__ idx, int* wsi) {
    int n = blockIdx.x * 256 + threadIdx.x;
    atomicAdd(&wsi[idx[n]], 1);
}
__global__ void sl_prefix(int* wsi) {
    if (threadIdx.x == 0) {
        int s = 0;
        for (int e = 0; e < NEXP; ++e) { wsi[8 + e] = s; wsi[16 + e] = s; s += wsi[e]; }
    }
}
__global__ void sl_scatter(const int* __restrict__ idx, int* wsi) {
    int n = blockIdx.x * 256 + threadIdx.x;
    int e = idx[n];
    int p = atomicAdd(&wsi[16 + e], 1);
    wsi[32 + p] = n;
}

// ---------------- device helpers ----------------
// Split an f32 pair into packed bf16 hi/lo pairs. Building v2bf vectors lets
// clang emit one two-source v_cvt_pk_bf16_f32 per pair (+ v_fma_mix for lo),
// instead of per-element cvt + v_mov_b16 merges.
__device__ __forceinline__ void pack_pair(float f0, float f1, uint32_t& ph, uint32_t& pl) {
    v2bf h = { (__bf16)f0, (__bf16)f1 };
    v2bf l = { (__bf16)(f0 - (float)h[0]), (__bf16)(f1 - (float)h[1]) };
    ph = __builtin_bit_cast(uint32_t, h);
    pl = __builtin_bit_cast(uint32_t, l);
}

// Load a v16bf operand as two 16B LDS reads; off2 = u32 distance to the second quad.
__device__ __forceinline__ v16bf load_frag(const uint32_t* p, int off2) {
    union { uint4 q[2]; v16bf v; } u;
    u.q[0] = *(const uint4*)(p);
    u.q[1] = *(const uint4*)(p + off2);
    return u.v;
}

// ---------------- grouped GEMM ----------------
__global__ void __launch_bounds__(THREADS)
sl_gemm(const float* __restrict__ x, const float* __restrict__ w,
        float* __restrict__ out, const int* __restrict__ wsi) {
    const int e = blockIdx.y >> 6;            // expert
    const int t = blockIdx.y & 63;            // row-tile slot within expert
    const int count = wsi[e];
    if (t * BM >= count) return;              // uniform early exit
    const int start = wsi[8 + e];
    const int rows_here = (count - t * BM < BM) ? (count - t * BM) : BM;
    const int gn0 = blockIdx.x * BN;

    // LDS: bf16 hi/lo tiles, packed K-pairs: T[row][kp] = {K=2kp, K=2kp+1}
    __shared__ uint32_t Ah[BM][PITCH], Al[BM][PITCH];   // A: [token-row][kpair]
    __shared__ uint32_t Bh[BN][PITCH], Bl[BN][PITCH];   // B: [out-col][kpair]
    __shared__ int      tok_s[BM];

    const int tid = threadIdx.x;
    if (tid < BM) {
        int g = t * BM + tid;
        tok_s[tid] = (g < count) ? wsi[32 + start + g] : -1;
    }
    __syncthreads();

    const int wave = tid >> 5, lane = tid & 31;
    const int wm = wave >> 2, wn = wave & 3;  // 2x4 wave grid -> 64x32 region each
    const int lhalf = (lane >> 4);            // 0: lanes 0-15, 1: lanes 16-31
    const int lrow  = lane & 15;

    v8f acc[4][2];
#pragma unroll
    for (int sm = 0; sm < 4; ++sm)
#pragma unroll
        for (int sn = 0; sn < 2; ++sn) acc[sm][sn] = (v8f){};

    // A loader: 2 threads/row, 16 consecutive f32 each (coalesced b128s)
    const int ar  = tid >> 1;
    const int akq = (tid & 1) * 16;           // f32 offset within chunk
    int atk = tok_s[ar]; if (atk < 0) atk = tok_s[0];
    const float* abase = x + (size_t)atk * D_IN + akq;

    // B loader: 1 thread per (col, 8 contiguous kpairs): n = tid&127, kp in [bc*8, bc*8+8)
    const int bn = tid & 127;
    const int bc = tid >> 7;                  // 0 or 1
    const float* bbase = w + (size_t)e * D_IN * D_OUTN + (size_t)(gn0 + bn);

    alignas(16) float a_reg[16];
    float b_reg[16];

    // ---- preload chunk 0 into registers ----
    {
        const float4* p = (const float4*)(abase);
#pragma unroll
        for (int i = 0; i < 4; ++i) *(float4*)(a_reg + 4 * i) = p[i];
        const size_t kb = (size_t)(bc * 16) * D_OUTN;
#pragma unroll
        for (int j = 0; j < 8; ++j) {
            b_reg[2 * j]     = bbase[kb + (size_t)(2 * j) * D_OUTN];
            b_reg[2 * j + 1] = bbase[kb + (size_t)(2 * j + 1) * D_OUTN];
        }
    }

    for (int kc = 0; kc < KCHUNKS; ++kc) {
        // ---- convert registers -> LDS (bf16 hi/lo, packed pairs) ----
        {
            alignas(16) uint32_t ph[8], pl[8];
#pragma unroll
            for (int j = 0; j < 8; ++j) pack_pair(a_reg[2 * j], a_reg[2 * j + 1], ph[j], pl[j]);
            const int kp0 = akq >> 1;         // 0 or 8
            *(uint4*)&Ah[ar][kp0]     = *(const uint4*)&ph[0];
            *(uint4*)&Ah[ar][kp0 + 4] = *(const uint4*)&ph[4];
            *(uint4*)&Al[ar][kp0]     = *(const uint4*)&pl[0];
            *(uint4*)&Al[ar][kp0 + 4] = *(const uint4*)&pl[4];
#pragma unroll
            for (int j = 0; j < 8; ++j) pack_pair(b_reg[2 * j], b_reg[2 * j + 1], ph[j], pl[j]);
            const int bkp0 = bc * 8;          // 0 or 8
            *(uint4*)&Bh[bn][bkp0]     = *(const uint4*)&ph[0];
            *(uint4*)&Bh[bn][bkp0 + 4] = *(const uint4*)&ph[4];
            *(uint4*)&Bl[bn][bkp0]     = *(const uint4*)&pl[0];
            *(uint4*)&Bl[bn][bkp0 + 4] = *(const uint4*)&pl[4];
        }
        __syncthreads();

        // ---- issue next chunk's global loads (latency hidden under WMMAs) ----
        if (kc + 1 < KCHUNKS) {
            const int k0n = (kc + 1) * BK;
            const float4* p = (const float4*)(abase + k0n);
#pragma unroll
            for (int i = 0; i < 4; ++i) *(float4*)(a_reg + 4 * i) = p[i];
            const size_t kb = (size_t)(k0n + bc * 16) * D_OUTN;
#pragma unroll
            for (int j = 0; j < 8; ++j) {
                b_reg[2 * j]     = bbase[kb + (size_t)(2 * j) * D_OUTN];
                b_reg[2 * j + 1] = bbase[kb + (size_t)(2 * j + 1) * D_OUTN];
            }
        }
        if (kc + 2 < KCHUNKS) {
            const int k0p = (kc + 2) * BK;
            __builtin_prefetch(abase + k0p, 0, 1);
            __builtin_prefetch(bbase + (size_t)(k0p + bc * 16) * D_OUTN, 0, 1);
            __builtin_prefetch(bbase + (size_t)(k0p + bc * 16 + 8) * D_OUTN, 0, 1);
        }

        // ---- fragments: pure b128 LDS reads, no conversion ----
        v16bf ah[4], al[4], bhf[2], blf[2];
        const int ako = lhalf * 4;            // A: kp {0..3,8..11} | {4..7,12..15}
        const int bko = lhalf * 4;            // B: kp {0..3,8..11} | {4..7,12..15}
#pragma unroll
        for (int sm = 0; sm < 4; ++sm) {
            int row = wm * 64 + sm * 16 + lrow;
            ah[sm] = load_frag(&Ah[row][ako], 8);
            al[sm] = load_frag(&Al[row][ako], 8);
        }
#pragma unroll
        for (int sn = 0; sn < 2; ++sn) {
            int col = wn * 32 + sn * 16 + lrow;
            bhf[sn] = load_frag(&Bh[col][bko], 8);
            blf[sn] = load_frag(&Bl[col][bko], 8);
        }

        // ---- 3-term split-bf16 WMMA: hi*hi + hi*lo + lo*hi ----
#pragma unroll
        for (int sm = 0; sm < 4; ++sm) {
#pragma unroll
            for (int sn = 0; sn < 2; ++sn) {
                acc[sm][sn] = __builtin_amdgcn_wmma_f32_16x16x32_bf16(
                    false, ah[sm], false, bhf[sn], (short)0, acc[sm][sn], false, false);
                acc[sm][sn] = __builtin_amdgcn_wmma_f32_16x16x32_bf16(
                    false, ah[sm], false, blf[sn], (short)0, acc[sm][sn], false, false);
                acc[sm][sn] = __builtin_amdgcn_wmma_f32_16x16x32_bf16(
                    false, al[sm], false, bhf[sn], (short)0, acc[sm][sn], false, false);
            }
        }
        __syncthreads();
    }

    // ---- epilogue: scatter rows through token list ----
#pragma unroll
    for (int sm = 0; sm < 4; ++sm) {
#pragma unroll
        for (int sn = 0; sn < 2; ++sn) {
            int col = gn0 + wn * 32 + sn * 16 + lrow;
            int rb  = wm * 64 + sm * 16 + lhalf * 8;
#pragma unroll
            for (int r = 0; r < 8; ++r) {
                int tr = rb + r;
                if (tr < rows_here) {
                    int tk = tok_s[tr];
                    out[(size_t)tk * D_OUTN + col] = acc[sm][sn][r];
                }
            }
        }
    }
}

// ---------------- launch ----------------
extern "C" void kernel_launch(void* const* d_in, const int* in_sizes, int n_in,
                              void* d_out, int out_size, void* d_ws, size_t ws_size,
                              hipStream_t stream) {
    const float* x   = (const float*)d_in[0];
    const float* w   = (const float*)d_in[1];
    const int*   idx = (const int*)d_in[2];
    float* out = (float*)d_out;
    int*   wsi = (int*)d_ws;

    sl_zero<<<1, 32, 0, stream>>>(wsi);
    sl_hist<<<N_TOK / 256, 256, 0, stream>>>(idx, wsi);
    sl_prefix<<<1, 32, 0, stream>>>(wsi);
    sl_scatter<<<N_TOK / 256, 256, 0, stream>>>(idx, wsi);

    dim3 grid(D_OUTN / BN, NEXP * (N_TOK / BM));
    sl_gemm<<<grid, THREADS, 0, stream>>>(x, w, out, wsi);
}
